// LSTMAggregator_49778670960869
// MI455X (gfx1250) — matrix operand order repaired and checked by reading, not compiled, not run on hardware
//
#include <hip/hip_runtime.h>
#include <hip/hip_bf16.h>

typedef __bf16 bf16;
typedef __attribute__((ext_vector_type(16))) __bf16 v16bf;
typedef __attribute__((ext_vector_type(8)))  float  v8f;

#define BB 128
#define TT 512
#define FF 256
#define HH 128
#define GG 512          // 4*H
#define HROW 136        // bank-conflict padded LDS row stride (bf16 elems)

union AFrag { v16bf v; uint4 q[2]; };
union BFrag { v16bf v; uint4 q[2]; };

// async-to-LDS builtin wants int4-vector pointers (per hipcc diagnostic)
typedef int v4i32 __attribute__((vector_size(16)));
typedef __attribute__((address_space(1))) v4i32* gas1_t;
typedef __attribute__((address_space(3))) v4i32* las3_t;

__device__ __forceinline__ bf16 f2bf(float f) {
    union { float f; unsigned u; } v; v.f = f;
    unsigned r = (v.u + 0x7FFFu + ((v.u >> 16) & 1u)) >> 16;
    union { unsigned short s; bf16 b; } o; o.s = (unsigned short)r; return o.b;
}
__device__ __forceinline__ float bf2f(bf16 b) {
    union { unsigned short s; bf16 b; } i; i.b = b;
    union { float f; unsigned u; } v; v.u = ((unsigned)i.s) << 16; return v.f;
}
__device__ __forceinline__ float sigf(float x) { return 1.0f / (1.0f + __expf(-x)); }

// ---- async global->LDS copy of one 16B chunk (CDNA5 ASYNCcnt path) ----
__device__ __forceinline__ void async_copy16(const bf16* g, bf16* l) {
#if __has_builtin(__builtin_amdgcn_global_load_async_to_lds_b128)
    __builtin_amdgcn_global_load_async_to_lds_b128(
        (gas1_t)(unsigned long long)(const void*)g,
        (las3_t)(unsigned)(unsigned long long)(void*)l, 0, 0);
#else
    *(uint4*)l = *(const uint4*)g;          // sync fallback
#endif
}
__device__ __forceinline__ void async_wait_all() {
#if __has_builtin(__builtin_amdgcn_s_wait_asynccnt)
    __builtin_amdgcn_s_wait_asynccnt(0);
#elif __has_builtin(__builtin_amdgcn_global_load_async_to_lds_b128)
    asm volatile("s_wait_asynccnt 0x0" ::: "memory");
#endif
}

// ---------------- fp32 -> bf16 elementwise (x) ----------------
__global__ void cvt_bf16_kernel(const float* __restrict__ src, bf16* __restrict__ dst, int n) {
    int i = (blockIdx.x * blockDim.x + threadIdx.x) * 4;
    if (i + 3 < n) {
        float4 f = *(const float4*)(src + i);
        ushort4 o;
        union { unsigned short s; bf16 b; } t;
        t.b = f2bf(f.x); o.x = t.s;
        t.b = f2bf(f.y); o.y = t.s;
        t.b = f2bf(f.z); o.z = t.s;
        t.b = f2bf(f.w); o.w = t.s;
        *(ushort4*)(dst + i) = o;
    }
}

// -------- transpose+convert: src [R,C] fp32 -> dst [C,R] bf16 --------
__global__ void transpose_bf16_kernel(const float* __restrict__ src, bf16* __restrict__ dst,
                                      int R, int C) {
    int idx = blockIdx.x * blockDim.x + threadIdx.x;
    if (idx < R * C) {
        int r = idx / C, c = idx % C;
        dst[(size_t)c * R + r] = f2bf(src[idx]);
    }
}

// ---------------- projection GEMM: xz[t,b,:] = x[b,t,:] @ W + bias ----------------
// A: [M=B*T, K=256] bf16 row-major.  Wt: [G=512, K] bf16 (W transposed).
// out: xz [T, B, G] bf16.  Block tile: 128(M) x 64(N), 8 waves (1 M-tile row each, 4 N-tiles).
__global__ __launch_bounds__(256)
void proj_gemm_kernel(const bf16* __restrict__ A, const bf16* __restrict__ Wt,
                      const float* __restrict__ bias, bf16* __restrict__ xz, int K) {
    extern __shared__ char smem[];
    const int KP = K + 8;                       // padded LDS stride
    bf16* lws = (bf16*)smem;                    // [64][KP]
    const int tid  = threadIdx.x;
    const int w    = tid >> 5;
    const int lane = tid & 31;
    const int l16  = lane & 15;
    const int hi   = lane >> 4;
    const int m0   = blockIdx.x * 128 + w * 16;
    const int n0   = blockIdx.y * 64;

    // stage 64 rows of Wt into LDS via async copies (16B chunks)
    for (int ci = tid; ci < 64 * (K / 8); ci += 256) {
        int r  = ci / (K / 8);
        int co = (ci % (K / 8)) * 8;
        async_copy16(Wt + (size_t)(n0 + r) * K + co, lws + (size_t)r * KP + co);
    }
    async_wait_all();
    __syncthreads();

    v8f acc0 = (v8f)(0.0f), acc1 = (v8f)(0.0f), acc2 = (v8f)(0.0f), acc3 = (v8f)(0.0f);
    for (int kb = 0; kb < K; kb += 32) {
        AFrag a;
        const bf16* ab = A + (size_t)(m0 + l16) * K + kb + hi * 16;
        a.q[0] = *(const uint4*)(ab);
        a.q[1] = *(const uint4*)(ab + 8);
        #pragma unroll
        for (int j = 0; j < 4; ++j) {
            BFrag bf;
            const bf16* bb = lws + (size_t)(j * 16 + l16) * KP + kb + hi * 16;
            bf.q[0] = ((const uint4*)bb)[0];
            bf.q[1] = ((const uint4*)bb)[1];
            v8f* accp = (j == 0) ? &acc0 : (j == 1) ? &acc1 : (j == 2) ? &acc2 : &acc3;
            *accp = __builtin_amdgcn_wmma_f32_16x16x32_bf16(
                false, a.v, false, bf.v, (short)0, *accp, false, false);
        }
    }

    // epilogue: +bias, bf16 store into [T,B,G]
    #pragma unroll
    for (int j = 0; j < 4; ++j) {
        v8f accv = (j == 0) ? acc0 : (j == 1) ? acc1 : (j == 2) ? acc2 : acc3;
        int col = n0 + j * 16 + l16;
        float bv = bias[col];
        #pragma unroll
        for (int r = 0; r < 8; ++r) {
            int row = m0 + r + 8 * hi;          // row = b*T + t
            int b_  = row / TT;
            int t_  = row % TT;
            xz[((size_t)t_ * BB + b_) * GG + col] = f2bf(accv[r] + bv);
        }
    }
}

// ---------------- recurrent scan (persistent over all T steps) ----------------
// xz: [T, B, G] bf16.  Ut: [G, H] bf16 (U transposed).
// hseq: [B, T, 2H] bf16 (layer-0 output) or nullptr.
// hlast: [B, 2H] fp32 (final output) or nullptr.
// Block = 256 thr (8 waves) handles 16 batch rows; wave w owns hidden cols [16w,16w+16).
// xz[t] slice (16 rows x 512 = 16KB contiguous) is double-buffered in LDS via
// async global->LDS loads: prefetch t+1 while computing step t.
__global__ __launch_bounds__(256)
void lstm_scan_kernel(const bf16* __restrict__ xz, const bf16* __restrict__ Ut,
                      bf16* __restrict__ hseq, float* __restrict__ hlast,
                      int reverse, int dirofs) {
    extern __shared__ char smem[];
    bf16* lu  = (bf16*)smem;                                  // Ut: [G][HROW]
    bf16* lh  = (bf16*)(smem + (size_t)GG * HROW * 2);        // h:  [16][HROW]
    bf16* lxz = (bf16*)(smem + ((size_t)GG * HROW + 16 * HROW) * 2); // [2][16*G]
    const int tid  = threadIdx.x;
    const int w    = tid >> 5;
    const int lane = tid & 31;
    const int l16  = lane & 15;
    const int hi   = lane >> 4;
    const int b0   = blockIdx.x * 16;
    const int ncol = w * 16;

    // stage full Ut (512x128 bf16 = 128KB) into WGP LDS once (padded rows)
    for (int ci = tid; ci < GG * (HH / 8); ci += 256) {
        int r  = ci >> 4;                          // HH/8 == 16 chunks per row
        int co = (ci & 15) * 8;
        async_copy16(Ut + (size_t)r * HH + co, lu + (size_t)r * HROW + co);
    }
    // zero h
    for (int i = tid; i < 16 * HROW; i += 256) lh[i] = f2bf(0.0f);

    // prefetch xz slice for first step into buffer 0 (contiguous 16KB)
    {
        int t0 = reverse ? (TT - 1) : 0;
        const bf16* src = xz + ((size_t)t0 * BB + b0) * GG;
        #pragma unroll
        for (int i = 0; i < 4; ++i) {
            int e = (tid + i * 256) * 8;
            async_copy16(src + e, lxz + e);
        }
    }
    async_wait_all();
    __syncthreads();

    v8f c_state = (v8f)(0.0f);

    for (int s = 0; s < TT; ++s) {
        const int t = reverse ? (TT - 1 - s) : s;
        const int buf = s & 1;
        bf16* zsrc = lxz + buf * (16 * GG);

        // prefetch next slice into the other buffer (its old contents were
        // fully consumed at step s-1; all waves are past that step's barrier)
        if (s + 1 < TT) {
            int tn = reverse ? (TT - 2 - s) : (s + 1);
            const bf16* src = xz + ((size_t)tn * BB + b0) * GG;
            bf16* dst = lxz + (1 - buf) * (16 * GG);
            #pragma unroll
            for (int i = 0; i < 4; ++i) {
                int e = (tid + i * 256) * 8;
                async_copy16(src + e, dst + e);
            }
        }

        // init z with hoisted projection xz[t] (from LDS stage)
        v8f z0, z1, z2, z3;
        #pragma unroll
        for (int r = 0; r < 8; ++r) {
            int rb = (r + 8 * hi) * GG + ncol + l16;
            z0[r] = bf2f(zsrc[rb + 0 * HH]);
            z1[r] = bf2f(zsrc[rb + 1 * HH]);
            z2[r] = bf2f(zsrc[rb + 2 * HH]);
            z3[r] = bf2f(zsrc[rb + 3 * HH]);
        }

        // z += h @ U  (A = h from LDS, B = Ut rows from LDS)
        #pragma unroll
        for (int kb = 0; kb < HH; kb += 32) {
            AFrag a;
            const bf16* ab = lh + (size_t)l16 * HROW + kb + hi * 16;
            a.q[0] = *(const uint4*)(ab);
            a.q[1] = *(const uint4*)(ab + 8);
            #pragma unroll
            for (int g = 0; g < 4; ++g) {
                BFrag bf;
                const bf16* bb = lu + (size_t)(g * HH + ncol + l16) * HROW + kb + hi * 16;
                bf.q[0] = ((const uint4*)bb)[0];
                bf.q[1] = ((const uint4*)bb)[1];
                v8f* zp = (g == 0) ? &z0 : (g == 1) ? &z1 : (g == 2) ? &z2 : &z3;
                *zp = __builtin_amdgcn_wmma_f32_16x16x32_bf16(
                    false, a.v, false, bf.v, (short)0, *zp, false, false);
            }
        }

        // gates + state update (c persists in fp32 registers)
        v8f hv;
        #pragma unroll
        for (int r = 0; r < 8; ++r) {
            float ig = sigf(z0[r]);
            float fg = sigf(z1[r]);
            float gg = tanhf(z2[r]);
            float og = sigf(z3[r]);
            float cn = fg * c_state[r] + ig * gg;
            c_state[r] = cn;
            hv[r] = og * tanhf(cn);
        }

        __syncthreads();   // all WMMA reads of old h complete
        {
            int col = ncol + l16;
            #pragma unroll
            for (int r = 0; r < 8; ++r) {
                int row = r + 8 * hi;
                bf16 hb = f2bf(hv[r]);
                lh[(size_t)row * HROW + col] = hb;
                if (hseq)
                    hseq[((size_t)(b0 + row) * TT + t) * (2 * HH) + dirofs + col] = hb;
            }
            if (hlast && s == TT - 1) {
                #pragma unroll
                for (int r = 0; r < 8; ++r) {
                    int row = r + 8 * hi;
                    hlast[(size_t)(b0 + row) * (2 * HH) + dirofs + col] = hv[r];
                }
            }
        }
        // wait for next-slice prefetch, then release everyone into step s+1
        async_wait_all();
        __syncthreads();
    }
}

extern "C" void kernel_launch(void* const* d_in, const int* in_sizes, int n_in,
                              void* d_out, int out_size, void* d_ws, size_t ws_size,
                              hipStream_t stream) {
    const float* x   = (const float*)d_in[0];
    const float* W0f = (const float*)d_in[1];
    const float* U0f = (const float*)d_in[2];
    const float* b0f = (const float*)d_in[3];
    const float* W0b = (const float*)d_in[4];
    const float* U0b = (const float*)d_in[5];
    const float* b0b = (const float*)d_in[6];
    const float* W1f = (const float*)d_in[7];
    const float* U1f = (const float*)d_in[8];
    const float* b1f = (const float*)d_in[9];
    const float* W1b = (const float*)d_in[10];
    const float* U1b = (const float*)d_in[11];
    const float* b1b = (const float*)d_in[12];
    float* out = (float*)d_out;

    const size_t BT = (size_t)BB * TT;
    char* ws = (char*)d_ws;
    bf16* x_bf = (bf16*)ws;                               // [B*T, 256]     32MB
    bf16* h0   = (bf16*)(ws + BT * FF * 2);               // [B, T, 256]    32MB
    bf16* xzf  = (bf16*)(ws + BT * FF * 2 + BT * 256 * 2);// [T, B, 512]    64MB
    bf16* xzb  = xzf + BT * GG;                           // [T, B, 512]    64MB
    bf16* wp   = xzb + BT * GG;
    bf16* W0ft = wp;              wp += (size_t)GG * FF;  // [512,256]
    bf16* U0ft = wp;              wp += (size_t)GG * HH;  // [512,128]
    bf16* W0bt = wp;              wp += (size_t)GG * FF;
    bf16* U0bt = wp;              wp += (size_t)GG * HH;
    bf16* W1ft = wp;              wp += (size_t)GG * FF;
    bf16* U1ft = wp;              wp += (size_t)GG * HH;
    bf16* W1bt = wp;              wp += (size_t)GG * FF;
    bf16* U1bt = wp;              wp += (size_t)GG * HH;

    // 1) convert x to bf16
    {
        int n = (int)(BT * FF);
        cvt_bf16_kernel<<<n / 4 / 256, 256, 0, stream>>>(x, x_bf, n);
    }
    // 2) transpose+convert weights: W [K,512]->[512,K], U [128,512]->[512,128]
    auto tr = [&](const float* s, bf16* d, int R, int C) {
        int n = R * C;
        transpose_bf16_kernel<<<(n + 255) / 256, 256, 0, stream>>>(s, d, R, C);
    };
    tr(W0f, W0ft, FF, GG); tr(U0f, U0ft, HH, GG);
    tr(W0b, W0bt, FF, GG); tr(U0b, U0bt, HH, GG);
    tr(W1f, W1ft, FF, GG); tr(U1f, U1ft, HH, GG);
    tr(W1b, W1bt, FF, GG); tr(U1b, U1bt, HH, GG);

    const size_t projLds = (size_t)64 * (FF + 8) * 2;              // 33792 B
    const size_t scanLds = ((size_t)GG * HROW + 16 * HROW) * 2     // Ut + h
                         + (size_t)2 * 16 * GG * 2;                // 2 xz stages
    dim3 pgrid((unsigned)(BT / 128), GG / 64);

    // 3) layer 0 projections
    proj_gemm_kernel<<<pgrid, 256, projLds, stream>>>(x_bf, W0ft, b0f, xzf, FF);
    proj_gemm_kernel<<<pgrid, 256, projLds, stream>>>(x_bf, W0bt, b0b, xzb, FF);
    // 4) layer 0 scans -> h0 sequence (bf16)
    lstm_scan_kernel<<<BB / 16, 256, scanLds, stream>>>(xzf, U0ft, h0, nullptr, 0, 0);
    lstm_scan_kernel<<<BB / 16, 256, scanLds, stream>>>(xzb, U0bt, h0, nullptr, 1, HH);
    // 5) layer 1 projections (input = h0, K = 256), reuse xz buffers
    proj_gemm_kernel<<<pgrid, 256, projLds, stream>>>(h0, W1ft, b1f, xzf, 2 * HH);
    proj_gemm_kernel<<<pgrid, 256, projLds, stream>>>(h0, W1bt, b1b, xzb, 2 * HH);
    // 6) layer 1 scans -> final hidden states into d_out [B, 2H] fp32
    lstm_scan_kernel<<<BB / 16, 256, scanLds, stream>>>(xzf, U1ft, nullptr, out, 0, 0);
    lstm_scan_kernel<<<BB / 16, 256, scanLds, stream>>>(xzb, U1bt, nullptr, out, 1, HH);
}
